// FullContrastiveLoss_49297634624124
// MI455X (gfx1250) — matrix operand order repaired and checked by reading, not compiled, not run on hardware
//
#include <hip/hip_runtime.h>
#include <hip/hip_bf16.h>

typedef _Float16 half_t;
typedef _Float16 v4h  __attribute__((ext_vector_type(4)));
typedef _Float16 v8h  __attribute__((ext_vector_type(8)));
typedef _Float16 v16h __attribute__((ext_vector_type(16)));
typedef float    v8f  __attribute__((ext_vector_type(8)));

#define NROWS 8192
#define DDIM  512
#define NCLS  16
#define INV_T2 20.0f
#define EPS_L 1e-5f
#define TILE_HALFS (16 * DDIM)   // one 16-row B tile in halves (16 KB)

// ---------------------------------------------------------------------------
// Kernel 1: row-normalize (torch F.normalize semantics) and convert to f16.
// grid = (N, 2), block = 128.  y=0 -> text, y=1 -> image.
// ---------------------------------------------------------------------------
__global__ __launch_bounds__(128) void normalize_f16_kernel(
    const float* __restrict__ text, const float* __restrict__ image,
    half_t* __restrict__ nt, half_t* __restrict__ ni) {
  const int row = blockIdx.x;
  const float* src = (blockIdx.y == 0) ? text : image;
  half_t* dst      = (blockIdx.y == 0) ? nt   : ni;
  const int tid = threadIdx.x;

  const float4 v = ((const float4*)(src + (size_t)row * DDIM))[tid];
  float s = v.x * v.x + v.y * v.y + v.z * v.z + v.w * v.w;

  #pragma unroll
  for (int m = 16; m >= 1; m >>= 1) s += __shfl_xor(s, m, 32);

  __shared__ float ws[4];
  if ((tid & 31) == 0) ws[tid >> 5] = s;
  __syncthreads();
  s = ws[0] + ws[1] + ws[2] + ws[3];

  const float nrm = sqrtf(s);
  const float scale = 1.0f / fmaxf(nrm, 1e-12f);

  v4h o;
  o.x = (half_t)(v.x * scale);
  o.y = (half_t)(v.y * scale);
  o.z = (half_t)(v.z * scale);
  o.w = (half_t)(v.w * scale);
  ((v4h*)(dst + (size_t)row * DDIM))[tid] = o;
}

// ---------------------------------------------------------------------------
// Async stage of rows 0..7 (8 x 512 halves) of a 16-row B tile into LDS (this thread's share:
// two 16-byte chunks). GVS addressing: vdst=LDS offset, vaddr=32-bit byte
// offset, saddr=64-bit base. Tracked by ASYNCcnt.
// ---------------------------------------------------------------------------
__device__ __forceinline__ void stage_tile_async(const half_t* __restrict__ F,
                                                 half_t* __restrict__ ldsBuf,
                                                 int j0, int tid) {
  #pragma unroll
  for (int c = tid; c < 512; c += 256) {
    const int brow = c >> 6;
    const int bcol = (c & 63) << 3;
    const unsigned lds_off =
        (unsigned)(size_t)(ldsBuf + brow * DDIM + bcol);
    const unsigned goff =
        (unsigned)(((unsigned)(j0 + brow) * DDIM + bcol) * sizeof(half_t));
    asm volatile("global_load_async_to_lds_b128 %0, %1, %2"
                 :
                 : "v"(lds_off), "v"(goff), "s"(F)
                 : "memory");
  }
}

// ---------------------------------------------------------------------------
// Kernel 2: WMMA Gram + streaming class-sum reduction + loss epilogue.
// grid = (N/128, 2), block = 256 (8 waves; wave w owns 16 rows).
// Double-buffered async global->LDS staging of B tiles.
// ---------------------------------------------------------------------------
__global__ __launch_bounds__(256, 1) void gram_loss_kernel(
    const half_t* __restrict__ nt, const half_t* __restrict__ ni,
    const int* __restrict__ label, const int* __restrict__ class_counts,
    float* __restrict__ out) {
  const half_t* __restrict__ F = (blockIdx.y == 0) ? nt : ni;

  const int tid      = threadIdx.x;
  const int wave     = tid >> 5;
  const int lane     = tid & 31;
  const int laneM    = lane & 15;
  const int laneHalf = lane >> 4;
  const int i0       = blockIdx.x * 128 + wave * 16;

  __shared__ alignas(32) half_t Bs[2 * TILE_HALFS];   // 32 KB double buffer
  __shared__ float accS[8][16][NCLS];
  __shared__ float redS[256];

  for (int t = tid; t < 8 * 16 * NCLS; t += 256) ((float*)accS)[t] = 0.0f;

  // A fragments for this wave's 16 rows, all 16 k-steps, VGPR-resident.
  v16h A[16];
  {
    const half_t* arow = F + (size_t)(i0 + laneM) * DDIM;
    #pragma unroll
    for (int kk = 0; kk < 16; ++kk) {
      A[kk] = *(const v16h*)(arow + kk * 32 + laneHalf * 16);
    }
  }

  // Prologue: stage tile 0 into buffer 0.
  stage_tile_async(F, Bs, 0, tid);
  asm volatile("s_wait_asynccnt 0x0" ::: "memory");
  __syncthreads();

  int   cls = 0;
  float strAcc[8];

  for (int jt = 0; jt < NROWS / 16; ++jt) {
    const int sIdx = jt >> 5;     // class stripe (512-wide)
    const int tIdx = jt & 31;     // tile within stripe
    const int buf  = jt & 1;

    if (tIdx == 0) {
      cls = label[sIdx * 512];
      #pragma unroll
      for (int r = 0; r < 8; ++r) strAcc[r] = 0.0f;
    }

    // Kick off async staging of the next tile into the other buffer.
    if (jt + 1 < NROWS / 16)
      stage_tile_async(F, Bs + (buf ^ 1) * TILE_HALFS, (jt + 1) * 16, tid);

    // 16 WMMAs on the current tile, two independent accumulator chains.
    v8f cA = {};
    v8f cB = {};
    const half_t* bbase = Bs + buf * TILE_HALFS + laneM * DDIM + laneHalf * 16;
    #pragma unroll
    for (int kk = 0; kk < 16; kk += 2) {
      v16h b0 = *(const v16h*)(bbase + kk * 32);
      v16h b1 = *(const v16h*)(bbase + kk * 32 + 32);
      cA = __builtin_amdgcn_wmma_f32_16x16x32_f16(
          false, A[kk], false, b0, (short)0, cA, false, false);
      cB = __builtin_amdgcn_wmma_f32_16x16x32_f16(
          false, A[kk + 1], false, b1, (short)0, cB, false, false);
    }

    // exp + accumulate; skipping the diagonal realizes "own - diag".
    const int j = jt * 16 + laneM;
    const int iBase = i0 + laneHalf * 8;
    #pragma unroll
    for (int r = 0; r < 8; ++r) {
      float e = __expf((cA[r] + cB[r]) * INV_T2);
      if (iBase + r == j) e = 0.0f;
      strAcc[r] += e;
    }

    if (tIdx == 31) {
      // cross-lane reduce over the 16 columns each lane covered
      #pragma unroll
      for (int r = 0; r < 8; ++r) {
        float v = strAcc[r];
        v += __shfl_xor(v, 1, 32);
        v += __shfl_xor(v, 2, 32);
        v += __shfl_xor(v, 4, 32);
        v += __shfl_xor(v, 8, 32);
        if (laneM == 0) accS[wave][laneHalf * 8 + r][cls] += v;
      }
    }

    // Drain this wave's async loads; barrier publishes LDS + retires buffer.
    asm volatile("s_wait_asynccnt 0x0" ::: "memory");
    __syncthreads();
  }

  // Epilogue: per-row loss, block reduce, one atomic per block.
  float contrib = 0.0f;
  if (tid < 128) {
    const int w = tid >> 4, rr = tid & 15;
    const int i = blockIdx.x * 128 + w * 16 + rr;
    const int ci = label[i];
    float total = 0.0f;
    #pragma unroll
    for (int c = 0; c < NCLS; ++c) total += accS[w][rr][c];
    const float own = accS[w][rr][ci];          // diagonal already excluded
    const float den = own + (total - own) * (1.0f / (NCLS - 1));
    const float pr  = -__logf(den / (den + EPS_L));
    contrib = pr / (float)class_counts[ci];
  }
  redS[tid] = contrib;
  __syncthreads();
  #pragma unroll
  for (int off = 128; off >= 1; off >>= 1) {
    if (tid < off) redS[tid] += redS[tid + off];
    __syncthreads();
  }
  if (tid == 0) atomicAdd(out, redS[0]);
}

// ---------------------------------------------------------------------------
extern "C" void kernel_launch(void* const* d_in, const int* in_sizes, int n_in,
                              void* d_out, int out_size, void* d_ws, size_t ws_size,
                              hipStream_t stream) {
  const float* text  = (const float*)d_in[0];
  const float* image = (const float*)d_in[1];
  const int* label   = (const int*)d_in[2];
  const int* counts  = (const int*)d_in[3];
  float* out         = (float*)d_out;

  half_t* nt = (half_t*)d_ws;                  // 8 MB
  half_t* ni = nt + (size_t)NROWS * DDIM;      // 8 MB

  hipMemsetAsync(d_out, 0, sizeof(float), stream);

  normalize_f16_kernel<<<dim3(NROWS, 2), 128, 0, stream>>>(text, image, nt, ni);
  gram_loss_kernel<<<dim3(NROWS / 128, 2), 256, 0, stream>>>(nt, ni, label,
                                                             counts, out);
}
